// BitLinearInference_61529701482550
// MI455X (gfx1250) — 8-bit matrix operand order repaired and checked by reading, compiled, not run on hardware
//
#include <hip/hip_runtime.h>

typedef __attribute__((ext_vector_type(8))) int v8i;

#define TOK    8192
#define IN_F   4096
#define OUT_F  16384
#define LDSS   80

__global__ __launch_bounds__(256)
void bl_quant(const float* __restrict__ x, const float* __restrict__ wscale,
              signed char* __restrict__ Aq, float* __restrict__ rowf) {
  const int row = blockIdx.x;
  const int t   = threadIdx.x;
  const float4* xr = (const float4*)(x + (size_t)row * IN_F);

  float4 v[4];
  float mx = 0.0f;
#pragma unroll
  for (int i = 0; i < 4; ++i) {
    v[i] = xr[t * 4 + i];
    mx = fmaxf(mx, fmaxf(fmaxf(fabsf(v[i].x), fabsf(v[i].y)),
                         fmaxf(fabsf(v[i].z), fabsf(v[i].w))));
  }
#pragma unroll
  for (int off = 16; off > 0; off >>= 1)
    mx = fmaxf(mx, __shfl_xor(mx, off, 32));
  __shared__ float red[8];
  if ((t & 31) == 0) red[t >> 5] = mx;
  __syncthreads();
  float m = red[0];
#pragma unroll
  for (int i = 1; i < 8; ++i) m = fmaxf(m, red[i]);

  const float s   = fmaxf(m, 1e-5f);
  const float inv = 127.0f / s;

  int pk[4];
#pragma unroll
  for (int i = 0; i < 4; ++i) {
    const int q0 = (int)rintf(fminf(fmaxf(v[i].x * inv, -128.0f), 127.0f));
    const int q1 = (int)rintf(fminf(fmaxf(v[i].y * inv, -128.0f), 127.0f));
    const int q2 = (int)rintf(fminf(fmaxf(v[i].z * inv, -128.0f), 127.0f));
    const int q3 = (int)rintf(fminf(fmaxf(v[i].w * inv, -128.0f), 127.0f));
    pk[i] = (q0 & 0xFF) | ((q1 & 0xFF) << 8) | ((q2 & 0xFF) << 16) | ((q3 & 0xFF) << 24);
  }
  int4 o; o.x = pk[0]; o.y = pk[1]; o.z = pk[2]; o.w = pk[3];
  ((int4*)(Aq + (size_t)row * IN_F))[t] = o;

  if (t == 0) rowf[row] = s * wscale[0] * (1.0f / 127.0f);
}

__global__ __launch_bounds__(256)
void bl_packw(const int4* __restrict__ wt, int* __restrict__ Bq) {
  const size_t i = (size_t)blockIdx.x * 256 + threadIdx.x;
  const int4 w = wt[i];
  Bq[i] = (w.x & 0xFF) | ((w.y & 0xFF) << 8) | ((w.z & 0xFF) << 16) | ((w.w & 0xFF) << 24);
}

__device__ __forceinline__ void bl_mma_step(const signed char* __restrict__ pa,
                                            const signed char* __restrict__ pb,
                                            int aBase, int bBase,
                                            v8i acc[4][2]) {
  v8i bf0, bf1;
  {
    const signed char* p = pb + bBase;
    const int4 e0 = *(const int4*)(p);
    const int4 e1 = *(const int4*)(p + 32);
    const v8i b = {e0.x, e0.y, e0.z, e0.w, e1.x, e1.y, e1.z, e1.w};
    bf0 = b;
  }
  {
    const signed char* p = pb + bBase + 16 * LDSS;
    const int4 e0 = *(const int4*)(p);
    const int4 e1 = *(const int4*)(p + 32);
    const v8i b = {e0.x, e0.y, e0.z, e0.w, e1.x, e1.y, e1.z, e1.w};
    bf1 = b;
  }
#pragma unroll
  for (int mi = 0; mi < 4; ++mi) {
    const signed char* p = pa + aBase + mi * (16 * LDSS);
    const int2 d0 = *(const int2*)(p);
    const int2 d1 = *(const int2*)(p + 8);
    const int2 d2 = *(const int2*)(p + 32);
    const int2 d3 = *(const int2*)(p + 40);
    const v8i a = {d0.x, d0.y, d1.x, d1.y, d2.x, d2.y, d3.x, d3.y};
    acc[mi][0] = __builtin_amdgcn_wmma_i32_16x16x64_iu8(
        true, a, true, bf0, acc[mi][0], false, false);
    acc[mi][1] = __builtin_amdgcn_wmma_i32_16x16x64_iu8(
        true, a, true, bf1, acc[mi][1], false, false);
  }
}

__global__ __launch_bounds__(256)
void bl_gemm(const signed char* __restrict__ Aq,
             const signed char* __restrict__ Bq,
             const float* __restrict__ rowf,
             float* __restrict__ out) {
  __shared__ signed char lA[2][128 * LDSS] __attribute__((aligned(16)));
  __shared__ signed char lB[2][128 * LDSS] __attribute__((aligned(16)));

  const int t    = threadIdx.x;
  const int lane = t & 31;
  const int wave = t >> 5;
  const int wm   = wave & 1;
  const int wn   = wave >> 1;
  const int m0   = blockIdx.y * 128;
  const int n0   = blockIdx.x * 128;

  const int grow = t >> 1;
  const int gcol = (t & 1) << 5;
  const signed char* gA = Aq + (size_t)(m0 + grow) * IN_F + gcol;
  const signed char* gB = Bq + (size_t)(n0 + grow) * IN_F + gcol;
  const int lofs = grow * LDSS + gcol;

  const int lrow  = lane & 15;
  const int ak0   = (lane >> 4) << 4;
  const int bk0   = (lane >> 4) << 4;
  const int aBase = (wm * 64 + lrow) * LDSS + ak0;
  const int bBase = (wn * 32 + lrow) * LDSS + bk0;

  const v8i vzero = {0, 0, 0, 0, 0, 0, 0, 0};
  v8i acc[4][2];
#pragma unroll
  for (int i = 0; i < 4; ++i)
#pragma unroll
    for (int j = 0; j < 2; ++j) acc[i][j] = vzero;

  int4 rA0 = *(const int4*)(gA);
  int4 rA1 = *(const int4*)(gA + 16);
  int4 rB0 = *(const int4*)(gB);
  int4 rB1 = *(const int4*)(gB + 16);
  *(int4*)(&lA[0][lofs]) = rA0;  *(int4*)(&lA[0][lofs + 16]) = rA1;
  *(int4*)(&lB[0][lofs]) = rB0;  *(int4*)(&lB[0][lofs + 16]) = rB1;

  const int NK = IN_F / 64;
#pragma unroll 1
  for (int kt = 0; kt < NK; kt += 2) {
    {
      const signed char* ga = gA + (kt + 1) * 64;
      const signed char* gb = gB + (kt + 1) * 64;
      rA0 = *(const int4*)(ga);
      rA1 = *(const int4*)(ga + 16);
      rB0 = *(const int4*)(gb);
      rB1 = *(const int4*)(gb + 16);
    }
    __syncthreads();
    bl_mma_step(&lA[0][0], &lB[0][0], aBase, bBase, acc);
    *(int4*)(&lA[1][lofs]) = rA0;  *(int4*)(&lA[1][lofs + 16]) = rA1;
    *(int4*)(&lB[1][lofs]) = rB0;  *(int4*)(&lB[1][lofs + 16]) = rB1;

    if (kt + 2 < NK) {
      const signed char* ga = gA + (kt + 2) * 64;
      const signed char* gb = gB + (kt + 2) * 64;
      rA0 = *(const int4*)(ga);
      rA1 = *(const int4*)(ga + 16);
      rB0 = *(const int4*)(gb);
      rB1 = *(const int4*)(gb + 16);
    }
    __syncthreads();
    bl_mma_step(&lA[1][0], &lB[1][0], aBase, bBase, acc);
    if (kt + 2 < NK) {
      *(int4*)(&lA[0][lofs]) = rA0;  *(int4*)(&lA[0][lofs + 16]) = rA1;
      *(int4*)(&lB[0][lofs]) = rB0;  *(int4*)(&lB[0][lofs + 16]) = rB1;
    }
  }

  const int mhalf = (lane >> 4) << 3;
  const int nn    = lane & 15;
#pragma unroll
  for (int mi = 0; mi < 4; ++mi) {
#pragma unroll
    for (int r = 0; r < 8; ++r) {
      const int m   = m0 + wm * 64 + mi * 16 + mhalf + r;
      const float f = rowf[m];
      float* o = out + (size_t)m * OUT_F + n0 + wn * 32 + nn;
      __builtin_nontemporal_store((float)acc[mi][0][r] * f, o);
      __builtin_nontemporal_store((float)acc[mi][1][r] * f, o + 16);
    }
  }
}

extern "C" void kernel_launch(void* const* d_in, const int* in_sizes, int n_in,
                              void* d_out, int out_size, void* d_ws, size_t ws_size,
                              hipStream_t stream) {
  (void)in_sizes; (void)n_in; (void)out_size; (void)ws_size;
  const float* x   = (const float*)d_in[0];
  const int*   wt  = (const int*)d_in[1];
  const float* wsc = (const float*)d_in[2];
  float* out = (float*)d_out;

  signed char* Aq   = (signed char*)d_ws;
  signed char* Bq   = Aq + (size_t)TOK * IN_F;
  float*       rowf = (float*)(Bq + (size_t)OUT_F * IN_F);

  bl_quant<<<TOK, 256, 0, stream>>>(x, wsc, Aq, rowf);
  bl_packw<<<(unsigned)(((size_t)OUT_F * IN_F) / 1024), 256, 0, stream>>>(
      (const int4*)wt, (int*)Bq);
  dim3 grid(OUT_F / 128, TOK / 128);
  bl_gemm<<<grid, 256, 0, stream>>>(Aq, Bq, rowf, out);
}
